// GlobalLossDminus_11501922419105
// MI455X (gfx1250) — compile-verified, matrix operand order checked by reading, not run on hardware
//
#include <hip/hip_runtime.h>
#include <hip/hip_bf16.h>

#define BATCH   4096
#define DIM     256
#define BS      (3 * BATCH)
#define LOG2E   1.4426950408889634f
#define LN2     0.6931471805599453f
#define T_LOG2E (10.0f * LOG2E)     // 1/temp * log2(e): exp(x/0.1) = exp2(x*T_LOG2E)
#define EPS     1e-6f
#define CSPLIT  8                   // column-dimension split for grid parallelism
#define CCHUNK  (BS / CSPLIT)       // 1536 columns per block

typedef __attribute__((ext_vector_type(16))) _Float16 v16h;
typedef __attribute__((ext_vector_type(8)))  _Float16 v8h;
typedef __attribute__((ext_vector_type(8)))  float    v8f;

// ---------------------------------------------------------------------------
// Kernel 1: cosine-normalize each row, write f16 copy for WMMA consumption.
// One wave32 per row; each lane handles 8 contiguous floats (256/32).
// ---------------------------------------------------------------------------
__global__ __launch_bounds__(256) void normalize_rows(
    const float* __restrict__ x, _Float16* __restrict__ xh)
{
    const int wave = (int)((blockIdx.x * blockDim.x + threadIdx.x) >> 5);
    const int lane = (int)(threadIdx.x & 31);
    if (wave >= BS) return;

    const float* row = x + (size_t)wave * DIM;
    float4 a = ((const float4*)row)[lane * 2];
    float4 b = ((const float4*)row)[lane * 2 + 1];

    float s = a.x * a.x + a.y * a.y + a.z * a.z + a.w * a.w
            + b.x * b.x + b.y * b.y + b.z * b.z + b.w * b.w;
    #pragma unroll
    for (int m = 16; m >= 1; m >>= 1) s += __shfl_xor(s, m, 32);

    const float inv = 1.0f / fmaxf(sqrtf(s), EPS);

    v8h o;
    o[0] = (_Float16)(a.x * inv); o[1] = (_Float16)(a.y * inv);
    o[2] = (_Float16)(a.z * inv); o[3] = (_Float16)(a.w * inv);
    o[4] = (_Float16)(b.x * inv); o[5] = (_Float16)(b.y * inv);
    o[6] = (_Float16)(b.z * inv); o[7] = (_Float16)(b.w * inv);
    *(v8h*)(xh + (size_t)wave * DIM + lane * 8) = o;
}

// ---------------------------------------------------------------------------
// Kernel 2: fused Gram + masked exp row-sum, M-blocked x2.
// Each wave owns 32 rows (two 16-row A tile sets, 128 VGPRs resident) and a
// 1536-column chunk. Per 16-column B tile: 8 k-steps; each B fragment feeds
// TWO chained v_wmma_f32_16x16x32_f16 (halves load bytes per FLOP vs x1).
// Epilogue per tile: exp2(c * 10*log2e) with compile-time lane mask, then
// accumulate; final 16-lane butterfly reduce writes 32 partial den values.
//
// Fragment layout per ISA 7.12.2 (16-bit, 16x32): lane l holds row/col (l&15);
// two b128 chunks at kBase + 8*(l>>4) and kBase + 8*(l>>4) + 16.
// C/D: element v of lane l is (M = v + 8*(l>>4), N = l&15).
// Mask: include iff col%4 != row%4  ->  (lane&3) != (v&3)  (tile bases %16==0,
// and the +16 row offset of the second M tile preserves row%4).
// ---------------------------------------------------------------------------
__global__ __launch_bounds__(256) void den_kernel(
    const _Float16* __restrict__ xh, float* __restrict__ denp)
{
    const int lane  = (int)(threadIdx.x & 31);
    const int wib   = (int)(threadIdx.x >> 5);
    const int rBase = (int)blockIdx.x * 256 + wib * 32;
    const int cbeg  = (int)blockIdx.y * CCHUNK;
    const int m16   = lane & 15;
    const int hi    = lane >> 4;
    const int khalf = hi * 8;

    // Resident A fragments: two row tiles, full K = 256 (8 k-steps of 32)
    v16h afrag0[8], afrag1[8];
    {
        const _Float16* ar0 = xh + (size_t)(rBase + m16) * DIM;
        const _Float16* ar1 = xh + (size_t)(rBase + 16 + m16) * DIM;
        #pragma unroll
        for (int ks = 0; ks < 8; ++ks) {
            union { v16h v; v8h h[2]; } u0, u1;
            u0.h[0] = *(const v8h*)(ar0 + ks * 32 + khalf);
            u0.h[1] = *(const v8h*)(ar0 + ks * 32 + khalf + 16);
            u1.h[0] = *(const v8h*)(ar1 + ks * 32 + khalf);
            u1.h[1] = *(const v8h*)(ar1 + ks * 32 + khalf + 16);
            afrag0[ks] = u0.v;
            afrag1[ks] = u1.v;
        }
    }

    float acc0[8], acc1[8];
    #pragma unroll
    for (int v = 0; v < 8; ++v) { acc0[v] = 0.0f; acc1[v] = 0.0f; }

    for (int cb = cbeg; cb < cbeg + CCHUNK; cb += 16) {
        const _Float16* brow = xh + (size_t)(cb + m16) * DIM;
        v8f c0 = {}, c1 = {};
        #pragma unroll
        for (int ks = 0; ks < 8; ++ks) {
            union { v16h v; v8h h[2]; } u;
            u.h[0] = *(const v8h*)(brow + ks * 32 + khalf);
            u.h[1] = *(const v8h*)(brow + ks * 32 + khalf + 16);
            c0 = __builtin_amdgcn_wmma_f32_16x16x32_f16(
                    false, afrag0[ks], false, u.v, (short)0, c0, false, false);
            c1 = __builtin_amdgcn_wmma_f32_16x16x32_f16(
                    false, afrag1[ks], false, u.v, (short)0, c1, false, false);
        }
        #pragma unroll
        for (int v = 0; v < 8; ++v) {
            const bool inc = (lane & 3) != (v & 3);
            float e0 = __builtin_amdgcn_exp2f(c0[v] * T_LOG2E);
            float e1 = __builtin_amdgcn_exp2f(c1[v] * T_LOG2E);
            acc0[v] += inc ? e0 : 0.0f;
            acc1[v] += inc ? e1 : 0.0f;
        }
    }

    // Butterfly-reduce across each 16-lane half; lane 0 -> row v, lane 16 -> row v+8
    float* dp = denp + (size_t)blockIdx.y * BS;
    #pragma unroll
    for (int v = 0; v < 8; ++v) {
        float s0 = acc0[v], s1 = acc1[v];
        s0 += __shfl_xor(s0, 1, 32); s1 += __shfl_xor(s1, 1, 32);
        s0 += __shfl_xor(s0, 2, 32); s1 += __shfl_xor(s1, 2, 32);
        s0 += __shfl_xor(s0, 4, 32); s1 += __shfl_xor(s1, 4, 32);
        s0 += __shfl_xor(s0, 8, 32); s1 += __shfl_xor(s1, 8, 32);
        if (m16 == 0) {
            dp[rBase + hi * 8 + v]      = s0;
            dp[rBase + 16 + hi * 8 + v] = s1;
        }
    }
}

// ---------------------------------------------------------------------------
// Kernel 3: deterministic reduction of the CSPLIT partial den slices.
// ---------------------------------------------------------------------------
__global__ __launch_bounds__(256) void den_reduce(
    const float* __restrict__ denp, float* __restrict__ den)
{
    const int k = (int)(blockIdx.x * 256 + threadIdx.x);
    if (k < BS) {
        float s = 0.0f;
        #pragma unroll
        for (int i = 0; i < CSPLIT; ++i) s += denp[(size_t)i * BS + k];
        den[k] = s;
    }
}

// ---------------------------------------------------------------------------
// Kernel 4: per-p pair losses. One wave per p; three 256-dim dots, then
// log2-space NT-Xent terms:  -log(n/(n+d)) = ln2*(log2(n+d) - log2(n)),
// and log2(n) == dot * 10 * log2(e) exactly.
// ---------------------------------------------------------------------------
__global__ __launch_bounds__(256) void pair_kernel(
    const _Float16* __restrict__ xh, const float* __restrict__ den,
    float* __restrict__ lossp)
{
    const int wave = (int)((blockIdx.x * blockDim.x + threadIdx.x) >> 5);
    const int lane = (int)(threadIdx.x & 31);
    if (wave >= BATCH) return;
    const int p = wave;

    const _Float16* r1 = xh + (size_t)p * DIM;
    const _Float16* r2 = xh + (size_t)(BATCH + p) * DIM;
    const _Float16* r3 = xh + (size_t)(2 * BATCH + p) * DIM;

    v8h a = *(const v8h*)(r1 + lane * 8);
    v8h b = *(const v8h*)(r2 + lane * 8);
    v8h c = *(const v8h*)(r3 + lane * 8);

    float d12 = 0.0f, d13 = 0.0f, d23 = 0.0f;
    #pragma unroll
    for (int i = 0; i < 8; ++i) {
        float x = (float)a[i], y = (float)b[i], z = (float)c[i];
        d12 += x * y; d13 += x * z; d23 += y * z;
    }
    #pragma unroll
    for (int m = 16; m >= 1; m >>= 1) {
        d12 += __shfl_xor(d12, m, 32);
        d13 += __shfl_xor(d13, m, 32);
        d23 += __shfl_xor(d23, m, 32);
    }

    if (lane == 0) {
        const float den1 = den[p];
        const float den2 = den[BATCH + p];
        const float den3 = den[2 * BATCH + p];
        const float l12 = d12 * T_LOG2E;   // log2(n12)
        const float l13 = d13 * T_LOG2E;
        const float l23 = d23 * T_LOG2E;
        const float n12 = __builtin_amdgcn_exp2f(l12);
        const float n13 = __builtin_amdgcn_exp2f(l13);
        const float n23 = __builtin_amdgcn_exp2f(l23);
        float t = 0.0f;
        t += __builtin_amdgcn_logf(n12 + den1) - l12;
        t += __builtin_amdgcn_logf(n12 + den2) - l12;
        t += __builtin_amdgcn_logf(n13 + den1) - l13;
        t += __builtin_amdgcn_logf(n13 + den3) - l13;
        t += __builtin_amdgcn_logf(n23 + den2) - l23;
        t += __builtin_amdgcn_logf(n23 + den3) - l23;
        lossp[p] = t * LN2;
    }
}

// ---------------------------------------------------------------------------
// Kernel 5: deterministic single-block reduction -> scalar output.
// ---------------------------------------------------------------------------
__global__ __launch_bounds__(256) void final_reduce(
    const float* __restrict__ lossp, float* __restrict__ out)
{
    __shared__ float sm[8];
    float s = 0.0f;
    for (int i = (int)threadIdx.x; i < BATCH; i += 256) s += lossp[i];
    #pragma unroll
    for (int m = 16; m >= 1; m >>= 1) s += __shfl_xor(s, m, 32);
    const int lane = (int)(threadIdx.x & 31), w = (int)(threadIdx.x >> 5);
    if (lane == 0) sm[w] = s;
    __syncthreads();
    if (threadIdx.x == 0) {
        float t = 0.0f;
        #pragma unroll
        for (int i = 0; i < 8; ++i) t += sm[i];
        out[0] = t / (2.0f * (float)BATCH);
    }
}

// ---------------------------------------------------------------------------
extern "C" void kernel_launch(void* const* d_in, const int* in_sizes, int n_in,
                              void* d_out, int out_size, void* d_ws, size_t ws_size,
                              hipStream_t stream)
{
    const float* x = (const float*)d_in[0];
    char* ws = (char*)d_ws;

    size_t off = 0;
    _Float16* xh    = (_Float16*)(ws + off); off += (size_t)BS * DIM * 2;      // 6,291,456 B
    float*    denp  = (float*)(ws + off);    off += (size_t)CSPLIT * BS * 4;   //   393,216 B
    float*    den   = (float*)(ws + off);    off += (size_t)BS * 4;            //    49,152 B
    float*    lossp = (float*)(ws + off);                                      //    16,384 B

    normalize_rows<<<BS / 8, 256, 0, stream>>>(x, xh);
    den_kernel    <<<dim3(BS / 256, CSPLIT), 256, 0, stream>>>(xh, denp);
    den_reduce    <<<BS / 256, 256, 0, stream>>>(denp, den);
    pair_kernel   <<<BATCH / 8, 256, 0, stream>>>(xh, den, lossp);
    final_reduce  <<<1, 256, 0, stream>>>(lossp, (float*)d_out);
}
